// DotProductAttention_44573170598739
// MI455X (gfx1250) — matrix operand order repaired and checked by reading, not compiled, not run on hardware
//
#include <hip/hip_runtime.h>

typedef __attribute__((ext_vector_type(16))) _Float16 v16h;
typedef __attribute__((ext_vector_type(8)))  float    v8f;

#define S_LEN 2048
#define D_DIM 64
#define BATCH 64
#define BM 128          // query rows per block (8 waves * 16)
#define BN 64           // key rows per tile iteration
#define NWAVE 8
#define NTILE (S_LEN / BN)
#define ELEMS ((BN * D_DIM) / 256)   // 16 elements per thread per tensor

// 1/sqrt(64) * log2(e): fold softmax scale + exp->exp2 conversion into Q
#define QSCALE (0.125f * 1.44269504088896f)

union FragU { v16h v; _Float16 h[16]; };

// A-matrix fragment (16-bit 16x32 layout, interleaved K per ISA table):
// lanes 0-15 (hi=0): halves 0..7 -> K=kb+0..7, halves 8..15 -> K=kb+16..23
// lanes 16-31 (hi=1): halves 0..7 -> K=kb+8..15, halves 8..15 -> K=kb+24..31
__device__ inline v16h frag_a_lds(const _Float16* row, int hi, int kb) {
  FragU u;
#pragma unroll
  for (int j = 0; j < 8; ++j) {
    u.h[j]     = row[kb + hi * 8 + j];
    u.h[j + 8] = row[kb + hi * 8 + 16 + j];
  }
  return u.v;
}

__device__ inline v16h frag_a_global_f32(const float* row, int hi, int kb, float scale) {
  FragU u;
#pragma unroll
  for (int j = 0; j < 8; ++j) {
    u.h[j]     = (_Float16)(row[kb + hi * 8 + j] * scale);
    u.h[j + 8] = (_Float16)(row[kb + hi * 8 + 16 + j] * scale);
  }
  return u.v;
}

// B-matrix fragment (column-per-lane, interleaved K matching A):
// lane holds column N=lane&15; halves 0..7 -> K=kb+hi*8+0..7, halves 8..15 -> K=kb+16+hi*8+0..7
__device__ inline v16h frag_b_lds(const _Float16* row, int hi, int kb) {
  FragU u;
#pragma unroll
  for (int j = 0; j < 8; ++j) {
    u.h[j]     = row[kb + hi * 8 + j];
    u.h[j + 8] = row[kb + hi * 8 + 16 + j];
  }
  return u.v;
}

__global__ __launch_bounds__(256)
void DotProductAttention_44573170598739_kernel(const float* __restrict__ Q,
                                               const float* __restrict__ K,
                                               const float* __restrict__ V,
                                               float* __restrict__ O) {
  __shared__ _Float16 Ks[BN][D_DIM];        // key tile, row-major (key, d)
  __shared__ _Float16 Vt[D_DIM][BN];        // value tile, transposed (d, key)
  __shared__ _Float16 Ps[NWAVE][16][BN];    // per-wave prob scratch (C->A relayout)

  const int tid  = threadIdx.x;
  const int wave = tid >> 5;
  const int lane = tid & 31;
  const int hi   = lane >> 4;
  const int ln   = lane & 15;
  const int b    = blockIdx.y;
  const int qbase = blockIdx.x * BM + wave * 16;

  // Q A-fragments (scaled into exp2 domain), reused across all key tiles
  const float* qrow = Q + ((size_t)b * S_LEN + qbase + ln) * D_DIM;
  const v16h qa0 = frag_a_global_f32(qrow, hi, 0,  QSCALE);
  const v16h qa1 = frag_a_global_f32(qrow, hi, 32, QSCALE);

  v8f o0 = {}, o1 = {}, o2 = {}, o3 = {};   // O accumulators, C layout, 4 d-subtiles
  float mrow[8], lrow[8];
#pragma unroll
  for (int r = 0; r < 8; ++r) { mrow[r] = -1e30f; lrow[r] = 0.0f; }

  const size_t kvbase = (size_t)b * S_LEN * D_DIM;

  // ---- software pipeline: registers hold the in-flight f32 tile ----
  float kreg[ELEMS], vreg[ELEMS];
#pragma unroll
  for (int i = 0; i < ELEMS; ++i) {
    size_t g = kvbase + (size_t)(tid + i * 256);
    kreg[i] = K[g];
    vreg[i] = V[g];
  }
  {
#pragma unroll
    for (int i = 0; i < ELEMS; ++i) {
      int idx = tid + i * 256;
      int r = idx >> 6, c = idx & 63;
      Ks[r][c] = (_Float16)kreg[i];
      Vt[c][r] = (_Float16)vreg[i];
    }
  }
  __syncthreads();

  for (int kt = 0; kt < NTILE; ++kt) {
    // ---- issue next tile's global loads (latency hidden under WMMA) ----
    const bool more = (kt + 1 < NTILE);
    if (more) {
      size_t tb = kvbase + (size_t)(kt + 1) * BN * D_DIM;
#pragma unroll
      for (int i = 0; i < ELEMS; ++i) {
        size_t g = tb + (size_t)(tid + i * 256);
        kreg[i] = K[g];
        vreg[i] = V[g];
      }
    }

    // ---- S = Q K^T : four 16-key subtiles, K-dim 64 = 2 x 32 ----
    v8f s0 = {}, s1 = {}, s2 = {}, s3 = {};
    {
      v16h kb00 = frag_b_lds(&Ks[ln][0],      hi, 0);
      v16h kb01 = frag_b_lds(&Ks[ln][0],      hi, 32);
      v16h kb10 = frag_b_lds(&Ks[16 + ln][0], hi, 0);
      v16h kb11 = frag_b_lds(&Ks[16 + ln][0], hi, 32);
      s0 = __builtin_amdgcn_wmma_f32_16x16x32_f16(false, qa0, false, kb00, (short)0, s0, false, false);
      s0 = __builtin_amdgcn_wmma_f32_16x16x32_f16(false, qa1, false, kb01, (short)0, s0, false, false);
      s1 = __builtin_amdgcn_wmma_f32_16x16x32_f16(false, qa0, false, kb10, (short)0, s1, false, false);
      s1 = __builtin_amdgcn_wmma_f32_16x16x32_f16(false, qa1, false, kb11, (short)0, s1, false, false);
      v16h kb20 = frag_b_lds(&Ks[32 + ln][0], hi, 0);
      v16h kb21 = frag_b_lds(&Ks[32 + ln][0], hi, 32);
      v16h kb30 = frag_b_lds(&Ks[48 + ln][0], hi, 0);
      v16h kb31 = frag_b_lds(&Ks[48 + ln][0], hi, 32);
      s2 = __builtin_amdgcn_wmma_f32_16x16x32_f16(false, qa0, false, kb20, (short)0, s2, false, false);
      s2 = __builtin_amdgcn_wmma_f32_16x16x32_f16(false, qa1, false, kb21, (short)0, s2, false, false);
      s3 = __builtin_amdgcn_wmma_f32_16x16x32_f16(false, qa0, false, kb30, (short)0, s3, false, false);
      s3 = __builtin_amdgcn_wmma_f32_16x16x32_f16(false, qa1, false, kb31, (short)0, s3, false, false);
    }

    // ---- online softmax in exp2 domain (scores already scaled by log2e/8) ----
    float alpha[8];
#pragma unroll
    for (int r = 0; r < 8; ++r) {
      float mx = fmaxf(fmaxf(s0[r], s1[r]), fmaxf(s2[r], s3[r]));
#pragma unroll
      for (int m = 1; m < 16; m <<= 1)
        mx = fmaxf(mx, __shfl_xor(mx, m, 32));
      float mnew = fmaxf(mrow[r], mx);
      float p0 = exp2f(s0[r] - mnew);
      float p1 = exp2f(s1[r] - mnew);
      float p2 = exp2f(s2[r] - mnew);
      float p3 = exp2f(s3[r] - mnew);
      float ssum = (p0 + p1) + (p2 + p3);
#pragma unroll
      for (int m = 1; m < 16; m <<= 1)
        ssum += __shfl_xor(ssum, m, 32);
      alpha[r] = exp2f(mrow[r] - mnew);
      lrow[r]  = lrow[r] * alpha[r] + ssum;
      mrow[r]  = mnew;
      s0[r] = p0; s1[r] = p1; s2[r] = p2; s3[r] = p3;
    }
#pragma unroll
    for (int r = 0; r < 8; ++r) {
      o0[r] *= alpha[r]; o1[r] *= alpha[r];
      o2[r] *= alpha[r]; o3[r] *= alpha[r];
    }

    // ---- P: C layout -> LDS -> A layout (per-wave scratch, wave-local) ----
#pragma unroll
    for (int r = 0; r < 8; ++r) {
      int M = hi ? (8 + r) : r;
      Ps[wave][M][ln]      = (_Float16)s0[r];
      Ps[wave][M][16 + ln] = (_Float16)s1[r];
      Ps[wave][M][32 + ln] = (_Float16)s2[r];
      Ps[wave][M][48 + ln] = (_Float16)s3[r];
    }
    v16h pa0 = frag_a_lds(&Ps[wave][ln][0], hi, 0);
    v16h pa1 = frag_a_lds(&Ps[wave][ln][0], hi, 32);

    // ---- O += P V : 4 output-d subtiles, K-dim 64 = 2 x 32 ----
    {
      v16h vb00 = frag_b_lds(&Vt[ln][0],      hi, 0);
      v16h vb01 = frag_b_lds(&Vt[ln][0],      hi, 32);
      v16h vb10 = frag_b_lds(&Vt[16 + ln][0], hi, 0);
      v16h vb11 = frag_b_lds(&Vt[16 + ln][0], hi, 32);
      o0 = __builtin_amdgcn_wmma_f32_16x16x32_f16(false, pa0, false, vb00, (short)0, o0, false, false);
      o0 = __builtin_amdgcn_wmma_f32_16x16x32_f16(false, pa1, false, vb01, (short)0, o0, false, false);
      o1 = __builtin_amdgcn_wmma_f32_16x16x32_f16(false, pa0, false, vb10, (short)0, o1, false, false);
      o1 = __builtin_amdgcn_wmma_f32_16x16x32_f16(false, pa1, false, vb11, (short)0, o1, false, false);
      v16h vb20 = frag_b_lds(&Vt[32 + ln][0], hi, 0);
      v16h vb21 = frag_b_lds(&Vt[32 + ln][0], hi, 32);
      v16h vb30 = frag_b_lds(&Vt[48 + ln][0], hi, 0);
      v16h vb31 = frag_b_lds(&Vt[48 + ln][0], hi, 32);
      o2 = __builtin_amdgcn_wmma_f32_16x16x32_f16(false, pa0, false, vb20, (short)0, o2, false, false);
      o2 = __builtin_amdgcn_wmma_f32_16x16x32_f16(false, pa1, false, vb21, (short)0, o2, false, false);
      o3 = __builtin_amdgcn_wmma_f32_16x16x32_f16(false, pa0, false, vb30, (short)0, o3, false, false);
      o3 = __builtin_amdgcn_wmma_f32_16x16x32_f16(false, pa1, false, vb31, (short)0, o3, false, false);
    }

    __syncthreads();   // all waves done reading this tile

    // ---- commit next tile to LDS ----
    if (more) {
#pragma unroll
      for (int i = 0; i < ELEMS; ++i) {
        int idx = tid + i * 256;
        int r = idx >> 6, c = idx & 63;
        Ks[r][c] = (_Float16)kreg[i];
        Vt[c][r] = (_Float16)vreg[i];
      }
    }
    __syncthreads();   // tile visible to all waves
  }

  // ---- epilogue: normalize and store fp32 ----
#pragma unroll
  for (int r = 0; r < 8; ++r) {
    int M = qbase + (hi ? (8 + r) : r);
    float inv = 1.0f / lrow[r];
    float* orow = O + ((size_t)b * S_LEN + M) * D_DIM;
    orow[ln]      = o0[r] * inv;
    orow[16 + ln] = o1[r] * inv;
    orow[32 + ln] = o2[r] * inv;
    orow[48 + ln] = o3[r] * inv;
  }
}

extern "C" void kernel_launch(void* const* d_in, const int* in_sizes, int n_in,
                              void* d_out, int out_size, void* d_ws, size_t ws_size,
                              hipStream_t stream) {
  const float* Q = (const float*)d_in[0];
  const float* K = (const float*)d_in[1];
  const float* V = (const float*)d_in[2];
  float* O = (float*)d_out;
  dim3 grid(S_LEN / BM, BATCH);
  DotProductAttention_44573170598739_kernel<<<grid, 256, 0, stream>>>(Q, K, V, O);
}
